// DeformableConv2d_695784702273
// MI455X (gfx1250) — matrix operand order repaired and checked by reading, not compiled, not run on hardware
//
#include <hip/hip_runtime.h>
#include <math.h>

typedef __attribute__((ext_vector_type(16))) __bf16 v16bf;
typedef __attribute__((ext_vector_type(8)))  float  v8f;

#define HWX   4096      // 64*64
#define CIN   256
#define COUT  256
#define KTAP  9
#define KSTEPS 72       // (9*256)/32

struct Samp { float w00, w01, w10, w11; int iy0, iy1, ix0, ix1; };

// ---------------------------------------------------------------------------
// Kernel 1: repack conv weights into WMMA A-fragment layout (bf16).
// A 16-bit 16x32 fragment: lane L (m = mt*16 + L%16), group g=L/16:
//   element e    -> k_local = g*16 + e
// Global K ordering: k = tap*256 + c.
// ---------------------------------------------------------------------------
__global__ __launch_bounds__(256)
void pack_weights_kernel(const float* __restrict__ weight,
                         const float* __restrict__ offw,
                         const float* __restrict__ modw,
                         __bf16* __restrict__ wpack,
                         __bf16* __restrict__ owpack) {
  int i = blockIdx.x * 256 + threadIdx.x;
  const int mainN = 16 * KSTEPS * 512;   // 589824 elements (256x2304)
  const int offN  = 2  * KSTEPS * 512;   // 73728  elements (32x2304)
  if (i >= mainN + offN) return;
  const bool isMain = i < mainN;
  int j  = isMain ? i : (i - mainN);
  int f  = j >> 9;          // fragment id
  int le = j & 511;
  int L  = le >> 4;         // lane
  int e  = le & 15;         // element within v16bf
  int mt = f / KSTEPS;
  int ks = f - mt * KSTEPS;
  int g  = L >> 4;
  int kl = g * 16 + e;
  int kg = ks * 32 + kl;
  int tap = kg >> 8;
  int c   = kg & 255;
  int m   = mt * 16 + (L & 15);
  if (isMain) {
    float v = weight[((size_t)m * CIN + c) * KTAP + tap];
    wpack[j] = (__bf16)v;
  } else {
    float v;
    if (m < 18)      v = offw[((size_t)m * CIN + c) * KTAP + tap];
    else if (m < 27) v = modw[((size_t)(m - 18) * CIN + c) * KTAP + tap];
    else             v = 0.f;
    owpack[j] = (__bf16)v;
  }
}

// ---------------------------------------------------------------------------
// Kernel 2: offset+mask conv as WMMA GEMM (M=32 padded, K=2304, N=32768).
// 128 threads = 4 waves; wave w owns pixel sub-tile w (16 pixels).
// ---------------------------------------------------------------------------
__global__ __launch_bounds__(128)
void offmask_kernel(const float* __restrict__ x, const v16bf* __restrict__ owp,
                    const float* __restrict__ offb, const float* __restrict__ modb,
                    float* __restrict__ param) {
  __shared__ v16bf s_b[2][4][32];
  const int tid  = threadIdx.x;
  const int lane = tid & 31;
  const int wv   = tid >> 5;          // wave = n-subtile
  const int pixBase = blockIdx.x * 64;

  // B writer: this thread fills all 16 elements of s_b[buf][bt][bL]
  const int bt = tid >> 5;
  const int bL = tid & 31;
  const int bn = bL & 15;
  const int khalf = bL >> 4;
  const int wpix = pixBase + bt * 16 + bn;
  const int wb   = wpix >> 12;
  const int whw  = wpix & 4095;
  const int oy = whw >> 6, ox = whw & 63;
  const float* xb = x + (size_t)wb * CIN * HWX + (size_t)(khalf * 16) * HWX;

  const v8f vzero = {};
  v8f acc0 = vzero, acc1 = vzero;
  int buf = 0;
  for (int tap = 0; tap < KTAP; ++tap) {
    // tap-invariant sampling setup (branch-free: clamp address, 0/1 mask)
    const int ki = tap / 3, kj = tap - 3 * (tap / 3);
    const int yy = oy - 1 + ki;
    const int xx = ox - 1 + kj;
    const float vm = (((unsigned)yy < 64u) && ((unsigned)xx < 64u)) ? 1.f : 0.f;
    const int iy = min(max(yy, 0), 63);
    const int ix = min(max(xx, 0), 63);
    const float* xs = xb + (iy * 64 + ix);
    for (int cb = 0; cb < 8; ++cb) {
      const int ks = tap * 8 + cb;
      const float* xc = xs + (size_t)(cb * 32) * HWX;
      union { uint4 u4[2]; __bf16 h[16]; } pk;
#pragma unroll
      for (int e = 0; e < 16; ++e)
        pk.h[e] = (__bf16)(xc[(size_t)e * HWX] * vm);
      uint4* dst = (uint4*)&s_b[buf][bt][bL];
      dst[0] = pk.u4[0];
      dst[1] = pk.u4[1];
      __syncthreads();
      v16bf a0 = owp[(size_t)(0 * KSTEPS + ks) * 32 + lane];
      v16bf a1 = owp[(size_t)(1 * KSTEPS + ks) * 32 + lane];
      v16bf bb = s_b[buf][wv][lane];
      acc0 = __builtin_amdgcn_wmma_f32_16x16x32_bf16(false, a0, false, bb,
                                                     (short)0, acc0, false, false);
      acc1 = __builtin_amdgcn_wmma_f32_16x16x32_bf16(false, a1, false, bb,
                                                     (short)0, acc1, false, false);
      buf ^= 1;
    }
  }

  // C/D layout: lane -> col n=lane%16, rows m = rh*8 + v (rh = lane/16).
  const int n  = lane & 15;
  const int rh = lane >> 4;
  const int pix = pixBase + wv * 16 + n;
#pragma unroll
  for (int v = 0; v < 8; ++v) {
    int m0 = rh * 8 + v;            // 0..15 -> always offset channels
    int m1 = 16 + rh * 8 + v;       // 16..31
    param[(size_t)pix * 32 + m0] = acc0[v] + offb[m0];
    if (m1 < 18) {
      param[(size_t)pix * 32 + m1] = acc1[v] + offb[m1];
    } else if (m1 < 27) {
      // 2*sigmoid via hardware v_rcp_f32 (avoids exact-division chain)
      float t = acc1[v] + modb[m1 - 18];
      param[(size_t)pix * 32 + m1] = 2.f * __builtin_amdgcn_rcpf(1.f + __expf(-t));
    }
  }
}

// ---------------------------------------------------------------------------
// Kernel 3: fused bilinear-sample + main GEMM.
// 256 threads = 8 waves; WG tile = 256 outputs x 64 pixels; wave = 32 rows.
// K=64 LDS panel per barrier (two 32-wide sub-panels, double buffered):
// 36 barriers total, 16 WMMAs + 4 A-fragment loads per sync window.
// ---------------------------------------------------------------------------
__global__ __launch_bounds__(256)
void deform_gemm_kernel(const float* __restrict__ x, const v16bf* __restrict__ wp,
                        const float* __restrict__ param, const float* __restrict__ bias,
                        float* __restrict__ out) {
  __shared__ v16bf s_b[2][2][4][32];     // [buf][sub][ntile][lane]  (16 KB)
  __shared__ Samp  s_samp[576];          // 64 pixels x 9 taps       (18 KB)
  const int tid  = threadIdx.x;
  const int lane = tid & 31;
  const int wv   = tid >> 5;
  const int pixBase = blockIdx.x * 64;

  // ---- stage 1: per-(pixel,tap) bilinear corners, mask-premultiplied ----
  for (int idx = tid; idx < 576; idx += 256) {
    int n   = idx / 9;
    int tap = idx - n * 9;
    int pix = pixBase + n;
    int hw  = pix & 4095;
    int oy = hw >> 6, ox = hw & 63;
    float dy = param[(size_t)pix * 32 + tap * 2 + 0];
    float dx = param[(size_t)pix * 32 + tap * 2 + 1];
    float mk = param[(size_t)pix * 32 + 18 + tap];
    int ki = tap / 3, kj = tap - 3 * (tap / 3);
    float py = (float)(oy - 1 + ki) + dy;
    float px = (float)(ox - 1 + kj) + dx;
    float fy0 = floorf(py), fx0 = floorf(px);
    float wy1 = py - fy0, wy0 = 1.f - wy1;
    float wx1 = px - fx0, wx0 = 1.f - wx1;
    int y0 = (int)fy0, x0 = (int)fx0;
    int y1 = y0 + 1,   x1 = x0 + 1;
    float vy0 = (y0 >= 0 && y0 <= 63) ? 1.f : 0.f;
    float vy1 = (y1 >= 0 && y1 <= 63) ? 1.f : 0.f;
    float vx0 = (x0 >= 0 && x0 <= 63) ? 1.f : 0.f;
    float vx1 = (x1 >= 0 && x1 <= 63) ? 1.f : 0.f;
    Samp s;
    s.w00 = wy0 * wx0 * mk * vy0 * vx0;
    s.w01 = wy0 * wx1 * mk * vy0 * vx1;
    s.w10 = wy1 * wx0 * mk * vy1 * vx0;
    s.w11 = wy1 * wx1 * mk * vy1 * vx1;
    s.iy0 = min(max(y0, 0), 63);
    s.iy1 = min(max(y1, 0), 63);
    s.ix0 = min(max(x0, 0), 63);
    s.ix1 = min(max(x1, 0), 63);
    s_samp[idx] = s;
  }
  __syncthreads();

  // ---- B writer constants: thread fills 8 bf16 per sub-panel lane ----
  const int bslot = tid * 8;             // element slot in 4x32x16 panel
  const int bt    = bslot >> 9;          // n-subtile
  const int bL    = (bslot >> 4) & 31;   // lane within fragment
  const int bn    = bL & 15;
  const int khalf = bL >> 4;
  const int e0    = (tid & 1) * 8;
  const int wpixn = bt * 16 + bn;        // pixel index within tile
  const int wpix  = pixBase + wpixn;
  const int wb    = wpix >> 12;
  const float* xb = x + (size_t)wb * CIN * HWX + (size_t)(khalf * 16 + e0) * HWX;

  const v8f vzero = {};
  v8f acc[2][4];
#pragma unroll
  for (int m = 0; m < 2; ++m)
#pragma unroll
    for (int t = 0; t < 4; ++t) acc[m][t] = vzero;

  const int mt0 = wv * 2;
  int buf = 0;
  for (int tap = 0; tap < KTAP; ++tap) {
    // tap-invariant bilinear descriptor (hoisted out of inner K loop)
    const Samp sp = s_samp[wpixn * 9 + tap];
    const int i00 = sp.iy0 * 64 + sp.ix0;
    const int i01 = sp.iy0 * 64 + sp.ix1;
    const int i10 = sp.iy1 * 64 + sp.ix0;
    const int i11 = sp.iy1 * 64 + sp.ix1;
    for (int cb2 = 0; cb2 < 4; ++cb2) {   // K=64 per iteration
#pragma unroll
      for (int sub = 0; sub < 2; ++sub) {
        const float* xc = xb + (size_t)((cb2 * 2 + sub) * 32) * HWX;
        union { uint4 u4; __bf16 h[8]; } pk;
#pragma unroll
        for (int j = 0; j < 8; ++j) {
          const float* xcj = xc + (size_t)j * HWX;
          float v = sp.w00 * xcj[i00] + sp.w01 * xcj[i01]
                  + sp.w10 * xcj[i10] + sp.w11 * xcj[i11];
          pk.h[j] = (__bf16)v;
        }
        *(uint4*)(((__bf16*)&s_b[buf][sub][bt][bL]) + e0) = pk.u4;
      }
      __syncthreads();

#pragma unroll
      for (int sub = 0; sub < 2; ++sub) {
        const int ks = tap * 8 + cb2 * 2 + sub;
        v16bf a0 = wp[((size_t)(mt0 + 0) * KSTEPS + ks) * 32 + lane];
        v16bf a1 = wp[((size_t)(mt0 + 1) * KSTEPS + ks) * 32 + lane];
#pragma unroll
        for (int t = 0; t < 4; ++t) {
          v16bf bb = s_b[buf][sub][t][lane];
          acc[0][t] = __builtin_amdgcn_wmma_f32_16x16x32_bf16(false, a0, false, bb,
                                                              (short)0, acc[0][t], false, false);
          acc[1][t] = __builtin_amdgcn_wmma_f32_16x16x32_bf16(false, a1, false, bb,
                                                              (short)0, acc[1][t], false, false);
        }
      }
      if (tap * 4 + cb2 + 1 < 36)
        __builtin_prefetch(&wp[((size_t)mt0 * KSTEPS + (tap * 8 + cb2 * 2 + 2)) * 32 + lane], 0, 3);
      buf ^= 1;
    }
  }

  // ---- store: lanes 0..15 = consecutive pixels (coalesced per channel) ----
  const int n  = lane & 15;
  const int rh = lane >> 4;
#pragma unroll
  for (int t = 0; t < 4; ++t) {
    int pix = pixBase + t * 16 + n;
    int b   = pix >> 12;
    int hw  = pix & 4095;
    float* ob = out + (size_t)b * COUT * HWX + hw;
#pragma unroll
    for (int ms = 0; ms < 2; ++ms) {
      int o0 = (mt0 + ms) * 16 + rh * 8;
#pragma unroll
      for (int v = 0; v < 8; ++v) {
        int o = o0 + v;
        ob[(size_t)o * HWX] = acc[ms][t][v] + bias[o];
      }
    }
  }
}

// ---------------------------------------------------------------------------
extern "C" void kernel_launch(void* const* d_in, const int* in_sizes, int n_in,
                              void* d_out, int out_size, void* d_ws, size_t ws_size,
                              hipStream_t stream) {
  const float* x      = (const float*)d_in[0];
  const float* offw   = (const float*)d_in[1];
  const float* offb   = (const float*)d_in[2];
  const float* modw   = (const float*)d_in[3];
  const float* modb   = (const float*)d_in[4];
  const float* weight = (const float*)d_in[5];
  const float* bias   = (const float*)d_in[6];
  float* out = (float*)d_out;

  char* ws = (char*)d_ws;
  __bf16* wpack  = (__bf16*)ws;                        // 1,179,648 B
  __bf16* owpack = (__bf16*)(ws + 1179648);            //   147,456 B
  float*  param  = (float*)(ws + 1179648 + 147456);    // 4,194,304 B

  pack_weights_kernel<<<2592, 256, 0, stream>>>(weight, offw, modw, wpack, owpack);
  offmask_kernel<<<512, 128, 0, stream>>>(x, (const v16bf*)owpack, offb, modb, param);
  deform_gemm_kernel<<<512, 256, 0, stream>>>(x, (const v16bf*)wpack, param, bias, out);
}
